// MaceNet_37787122270956
// MI455X (gfx1250) — matrix operand order repaired and checked by reading, not compiled, not run on hardware
//
#include <hip/hip_runtime.h>
#include <hip/hip_bf16.h>
#include <math.h>

// ---------------- Types for CDNA5 WMMA ----------------
typedef __bf16 bf16;
typedef __attribute__((ext_vector_type(16))) __bf16 v16bf;
typedef __attribute__((ext_vector_type(8)))  float  v8f;

// ---------------- Problem constants -------------------
#define L_LAYERS 2
#define F0 64
#define F1 32
#define NBASIS 8
#define NINV 64
#define NVEC 16
#define BATCH 4
#define NNODES 512
#define NEDGES (NNODES * (NNODES - 1))   // 261632
#define NTILES (NEDGES / 16)             // 16352 (exact)
#define INV_AVG (1.0f / (float)NNODES)

#define WPB 4  // waves per block in the edge kernel (128 threads)

// Per-layer bf16 fragment buffer layout (element offsets)
#define FR_WR1 0        // 32x64  (K padded 8->32) : 2048
#define FR_WR2 2048     // 64x192                  : 12288
#define FR_WM  14336    // 64x64                   : 4096
#define FR_WSV 18432    // 64x32                   : 2048
#define FR_WVV 20480    // 32x32                   : 1024
#define FR_WV0 21504    // 32x64                   : 2048
#define FR_LAYER 23552

// Intra-wave LDS ordering (CDNA5 split counters)
#define WAVE_SYNC() asm volatile("s_wait_dscnt 0" ::: "memory")

// Fast exp via the TRANS pipe (single v_exp_f32; co-executes with WMMA)
__device__ __forceinline__ float fast_exp(float x) { return __expf(x); }
__device__ __forceinline__ float silu_f(float x) { return x / (1.0f + __expf(-x)); }

__device__ __forceinline__ v8f wmma_bf16(v16bf a, v16bf b, v8f c) {
  // D = A(16x32 bf16) * B(32x16 bf16) + C(16x16 f32)
  return __builtin_amdgcn_wmma_f32_16x16x32_bf16(false, a, false, b, (short)0, c,
                                                 false, false);
}

// Build a 16x32 bf16 A-fragment from an LDS row-major tile [16][ldk].
// Per ISA: lane<16 -> M=lane, K in {0..7, 16..23}; lane>=16 -> K in {8..15, 24..31}.
// 8 contiguous bf16 per half => lowers to 2x ds_load_b128.
__device__ __forceinline__ v16bf load_a(const bf16* tile, int ldk, int k0, int lane) {
  v16bf a;
  const int m  = lane & 15;
  const int g8 = (lane >> 4) << 3;
  const bf16* row = tile + m * ldk + k0 + g8;
#pragma unroll
  for (int i = 0; i < 8; ++i) { a[i] = row[i]; a[i + 8] = row[i + 16]; }
  return a;
}

// B-fragment: pre-permuted in global memory, one contiguous 32B vector per lane.
__device__ __forceinline__ v16bf load_b(const bf16* wf, int nbt, int kb, int nb, int lane) {
  return *(const v16bf*)(wf + (((kb * nbt + nb) * 32) + lane) * 16);
}

// ---------------- Weight prep: f32 -> bf16 B-fragment layout ----------------
__global__ void prep_weights_kernel(const float* __restrict__ Wr1, const float* __restrict__ Wr2,
                                    const float* __restrict__ Wm,  const float* __restrict__ Wsv,
                                    const float* __restrict__ Wvv, const float* __restrict__ Wv0,
                                    bf16* __restrict__ wfrag) {
  const int bid  = blockIdx.x;
  const int l    = bid / 6;
  const int widx = bid % 6;
  const int Ks[6]   = {8, 64, 64, 64, 32, 32};
  const int Ns[6]   = {64, 192, 64, 32, 32, 64};
  const int Kp[6]   = {32, 64, 64, 64, 32, 32};
  const int dOff[6] = {FR_WR1, FR_WR2, FR_WM, FR_WSV, FR_WVV, FR_WV0};
  const float* srcs[6] = {Wr1 + l * 8 * 64,  Wr2 + l * 64 * 192, Wm + l * 64 * 64,
                          Wsv + l * 64 * 32, Wvv + l * 32 * 32,  Wv0 + l * 32 * 64};
  const int K = Ks[widx], N = Ns[widx], Kpad = Kp[widx];
  const int nbt = N / 16;
  const int total = (Kpad / 32) * nbt * 512;
  const float* src = srcs[widx];
  bf16* dst = wfrag + l * FR_LAYER + dOff[widx];
  for (int idx = threadIdx.x; idx < total; idx += blockDim.x) {
    int frag = idx >> 9;
    int lane = (idx >> 4) & 31;
    int i    = idx & 15;
    int kb = frag / nbt, nb = frag % nbt;
    int k = kb * 32 + ((lane >> 4) << 3) + (i & 7) + ((i >> 3) << 4);   // K = kb*32 + g*8 + (i&7) + (i>>3)*16
    int n = nb * 16 + (lane & 15);
    float v = (k < K) ? src[k * N + n] : 0.0f;
    dst[idx] = (bf16)v;
  }
}

// ---------------- State init: h_s = embed broadcast ----------------
__global__ void init_state_kernel(const float* __restrict__ embed, float* __restrict__ h_s) {
  int tid = blockIdx.x * blockDim.x + threadIdx.x;
  if (tid < BATCH * NNODES * F0) h_s[tid] = embed[tid & (F0 - 1)];
}

// ---------------- Edge kernel: WMMA message passing ----------------
__global__ void __launch_bounds__(WPB * 32)
edge_kernel(const float* __restrict__ x, const int* __restrict__ senders,
            const int* __restrict__ receivers, const float* __restrict__ h_s,
            const float* __restrict__ h_v, const bf16* __restrict__ wfrag,
            float* __restrict__ a0, float* __restrict__ a1, int layer) {
  __shared__ bf16  s_hsA[WPB][16 * F0];      // sender h_s tile (A-operand staging)
  __shared__ bf16  s_t1 [WPB][16 * 64];      // silu(rb@Wr1) re-fragment buffer
  __shared__ bf16  s_hv [WPB][3][16 * F1];   // sender h_v per component
  __shared__ bf16  s_hvu[WPB][16 * F1];      // hv . u
  __shared__ bf16  s_rb [WPB][16 * 32];      // radial basis (K padded to 32)
  __shared__ float s_u  [WPB][16][3];
  __shared__ int   s_si [WPB][16];
  __shared__ int   s_ri [WPB][16];

  const int w    = threadIdx.x >> 5;
  const int lane = threadIdx.x & 31;
  const int b    = blockIdx.y;

  const float* xb  = x   + b * NNODES * 3;
  const float* hsb = h_s + b * NNODES * F0;
  const float* hvb = h_v + b * NNODES * F1 * 3;
  float* a0b = a0 + b * NNODES * F0;
  float* a1b = a1 + b * NNODES * F1 * 3;

  const bf16* wl   = wfrag + layer * FR_LAYER;
  const bf16* fWr1 = wl + FR_WR1;
  const bf16* fWr2 = wl + FR_WR2;
  const bf16* fWm  = wl + FR_WM;
  const bf16* fWsv = wl + FR_WSV;
  const bf16* fWvv = wl + FR_WVV;
  const bf16* fWv0 = wl + FR_WV0;

  const int waveStride = gridDim.x * WPB;
  for (int tile = blockIdx.x * WPB + w; tile < NTILES; tile += waveStride) {
    // ---- Phase A: per-edge geometry (one edge per lane, lanes 0..15) ----
    if (lane < 16) {
      const int e = tile * 16 + lane;
      const int s = senders[e];
      const int r = receivers[e];
      s_si[w][lane] = s;
      s_ri[w][lane] = r;
      __builtin_prefetch(hsb + s * F0, 0, 3);       // global_prefetch_b8
      float vx = xb[r * 3 + 0] - xb[s * 3 + 0];
      float vy = xb[r * 3 + 1] - xb[s * 3 + 1];
      float vz = xb[r * 3 + 2] - xb[s * 3 + 2];
      float d  = sqrtf(vx * vx + vy * vy + vz * vz);
      float iv = 1.0f / (d + 1e-8f);
      s_u[w][lane][0] = vx * iv;
      s_u[w][lane][1] = vy * iv;
      s_u[w][lane][2] = vz * iv;
#pragma unroll
      for (int j = 0; j < NBASIS; ++j) {
        float c = 4.0f * (float)j / 7.0f;           // linspace(0,4,8)
        float t = d - c;
        s_rb[w][lane * 32 + j] = (bf16)fast_exp(-4.0f * t * t);
      }
#pragma unroll
      for (int j = NBASIS; j < 32; ++j) s_rb[w][lane * 32 + j] = (bf16)0.0f;
    }
    WAVE_SYNC();

    // ---- Phase B: gather sender features into bf16 LDS tiles ----
#pragma unroll 4
    for (int j = 0; j < 32; ++j) {                  // 16x64 hs tile
      int fidx = lane + 32 * j;
      int m = fidx >> 6, k = fidx & 63;
      s_hsA[w][m * F0 + k] = (bf16)hsb[s_si[w][m] * F0 + k];
    }
#pragma unroll 4
    for (int j = 0; j < 48; ++j) {                  // 3 x (16x32) hv tiles
      int fidx = lane + 32 * j;
      int i = fidx >> 9;
      int rem = fidx & 511;
      int m = rem >> 5, k = rem & 31;
      s_hv[w][i][m * F1 + k] = (bf16)hvb[(s_si[w][m] * F1 + k) * 3 + i];
    }
    WAVE_SYNC();
#pragma unroll 4
    for (int j = 0; j < 16; ++j) {                  // hv . u (16x32)
      int fidx = lane + 32 * j;
      int m = fidx >> 5, k = fidx & 31;
      float acc = (float)s_hv[w][0][m * F1 + k] * s_u[w][m][0]
                + (float)s_hv[w][1][m * F1 + k] * s_u[w][m][1]
                + (float)s_hv[w][2][m * F1 + k] * s_u[w][m][2];
      s_hvu[w][m * F1 + k] = (bf16)acc;
    }
    WAVE_SYNC();

    // ---- Phase C1: t1 = silu(rb @ Wr1), restaged D-layout -> A-layout ----
    {
      v16bf aRB = load_a(s_rb[w], 32, 0, lane);
      const int mBase = (lane >> 4) * 8;
      const int nl    = lane & 15;
#pragma unroll
      for (int nt = 0; nt < 4; ++nt) {
        v8f acc = {};
        acc = wmma_bf16(aRB, load_b(fWr1, 4, 0, nt, lane), acc);
#pragma unroll
        for (int r = 0; r < 8; ++r)
          s_t1[w][(mBase + r) * 64 + nt * 16 + nl] = (bf16)silu_f(acc[r]);
      }
    }
    WAVE_SYNC();

    // Hoisted A-fragments (reused across all output tiles)
    v16bf aHS0 = load_a(s_hsA[w], F0, 0, lane);
    v16bf aHS1 = load_a(s_hsA[w], F0, 32, lane);
    v16bf aT10 = load_a(s_t1[w], 64, 0, lane);
    v16bf aT11 = load_a(s_t1[w], 64, 32, lane);
    v16bf aHVU = load_a(s_hvu[w], F1, 0, lane);

    const int g  = lane >> 4;
    const int nl = lane & 15;

    // ---- Phase C2: m0 = w0*(hs@Wm) + w3*((hv.u)@Wv0) -> a0 (atomics) ----
#pragma unroll
    for (int nt = 0; nt < 4; ++nt) {
      v8f accM = {};
      accM = wmma_bf16(aHS0, load_b(fWm, 4, 0, nt, lane), accM);
      accM = wmma_bf16(aHS1, load_b(fWm, 4, 1, nt, lane), accM);
      v8f accV = {};
      accV = wmma_bf16(aHVU, load_b(fWv0, 4, 0, nt, lane), accV);
      v8f w0a = {};
      w0a = wmma_bf16(aT10, load_b(fWr2, 12, 0, nt, lane), w0a);
      w0a = wmma_bf16(aT11, load_b(fWr2, 12, 1, nt, lane), w0a);
      v8f w3a = {};
      w3a = wmma_bf16(aT10, load_b(fWr2, 12, 0, 4 + nt, lane), w3a);
      w3a = wmma_bf16(aT11, load_b(fWr2, 12, 1, 4 + nt, lane), w3a);
      const int col = nt * 16 + nl;
#pragma unroll
      for (int r = 0; r < 8; ++r) {
        int m = r + 8 * g;
        float val = (w0a[r] * accM[r] + w3a[r] * accV[r]) * INV_AVG;
        atomicAdd(&a0b[s_ri[w][m] * F0 + col], val);
      }
    }

    // ---- Phase C3: m1 = w1*(hs@Wsv)*u + w2*(T . (hv@Wvv)) -> a1 ----
#pragma unroll
    for (int nt = 0; nt < 2; ++nt) {
      v8f accS = {};
      accS = wmma_bf16(aHS0, load_b(fWsv, 2, 0, nt, lane), accS);
      accS = wmma_bf16(aHS1, load_b(fWsv, 2, 1, nt, lane), accS);
      v8f w1a = {};
      w1a = wmma_bf16(aT10, load_b(fWr2, 12, 0, 8 + nt, lane), w1a);
      w1a = wmma_bf16(aT11, load_b(fWr2, 12, 1, 8 + nt, lane), w1a);
      v8f w2a = {};
      w2a = wmma_bf16(aT10, load_b(fWr2, 12, 0, 10 + nt, lane), w2a);
      w2a = wmma_bf16(aT11, load_b(fWr2, 12, 1, 10 + nt, lane), w2a);
      v16bf bVV = load_b(fWvv, 2, 0, nt, lane);
      v8f mix0 = {}, mix1 = {}, mix2 = {};
      mix0 = wmma_bf16(load_a(s_hv[w][0], F1, 0, lane), bVV, mix0);
      mix1 = wmma_bf16(load_a(s_hv[w][1], F1, 0, lane), bVV, mix1);
      mix2 = wmma_bf16(load_a(s_hv[w][2], F1, 0, lane), bVV, mix2);
      const int col = nt * 16 + nl;
#pragma unroll
      for (int r = 0; r < 8; ++r) {
        int m = r + 8 * g;
        float u0 = s_u[w][m][0], u1 = s_u[w][m][1], u2 = s_u[w][m][2];
        float udm = u0 * mix0[r] + u1 * mix1[r] + u2 * mix2[r];
        float sw  = w1a[r] * accS[r];
        float* dst = &a1b[(s_ri[w][m] * F1 + col) * 3];
        float v0 = (sw * u0 + w2a[r] * (u0 * udm - mix0[r] * (1.0f / 3.0f))) * INV_AVG;
        float v1 = (sw * u1 + w2a[r] * (u1 * udm - mix1[r] * (1.0f / 3.0f))) * INV_AVG;
        float v2 = (sw * u2 + w2a[r] * (u2 * udm - mix2[r] * (1.0f / 3.0f))) * INV_AVG;
        atomicAdd(dst + 0, v0);
        atomicAdd(dst + 1, v1);
        atomicAdd(dst + 2, v2);
      }
    }
    WAVE_SYNC();
  }
}

// ---------------- Node update: h_s = h_s@Wself + corr(a0); h_v += a1 ----------------
__global__ void node_update_kernel(const float* __restrict__ Wc, const float* __restrict__ Wself,
                                   float* __restrict__ h_s, float* __restrict__ h_v,
                                   const float* __restrict__ a0, const float* __restrict__ a1,
                                   int layer) {
  const int node = blockIdx.x;   // b*N + n
  const int f    = threadIdx.x;  // 0..63
  __shared__ float sa0[F0], shs[F0];
  const float* a0r = a0 + node * F0;
  float* hsr = h_s + node * F0;
  sa0[f] = a0r[f];
  shs[f] = hsr[f];
  __syncthreads();
  const float* Wc0 = Wc + layer * 3 * F0 * F0;
  const float* Wc1 = Wc0 + F0 * F0;
  const float* Wc2 = Wc1 + F0 * F0;
  const float* Ws  = Wself + layer * F0 * F0;
  float acc = 0.0f;
#pragma unroll 4
  for (int k = 0; k < F0; ++k) {
    float a = sa0[k], a2 = a * a, a3 = a2 * a;
    acc += a * Wc0[k * F0 + f] + a2 * Wc1[k * F0 + f] + a3 * Wc2[k * F0 + f]
         + shs[k] * Ws[k * F0 + f];
  }
  __syncthreads();
  hsr[f] = acc;
  float* hvr = h_v + node * F1 * 3;
  const float* a1r = a1 + node * F1 * 3;
  for (int j = f; j < F1 * 3; j += F0) hvr[j] += a1r[j];
}

// ---------------- Output: inv = silu(h_s)@Wouts@Wfin + b; vecf = hv.Woutv ----------------
__global__ void output_kernel(const float* __restrict__ h_s, const float* __restrict__ h_v,
                              const float* __restrict__ Wouts, const float* __restrict__ Woutv,
                              const float* __restrict__ Wfin, const float* __restrict__ bfin,
                              float* __restrict__ out_vec, float* __restrict__ out_inv) {
  const int node = blockIdx.x;
  const int t    = threadIdx.x;  // 0..63
  __shared__ float s1[NINV], s2[NINV];
  float hs = h_s[node * F0 + t];
  s1[t] = hs / (1.0f + __expf(-hs));
  __syncthreads();
  float acc = 0.0f;
#pragma unroll 4
  for (int k = 0; k < F0; ++k) acc += s1[k] * Wouts[k * NINV + t];
  s2[t] = acc;
  __syncthreads();
  float acc2 = 0.0f;
#pragma unroll 4
  for (int k = 0; k < NINV; ++k) acc2 += s2[k] * Wfin[k * NINV + t];
  out_inv[node * NINV + t] = acc2 + bfin[t];
  const float* hv = h_v + node * F1 * 3;
  for (int idx = t; idx < NVEC * 3; idx += F0) {
    int gg = idx / 3, ii = idx % 3;
    float a = 0.0f;
#pragma unroll 4
    for (int ff = 0; ff < F1; ++ff) a += hv[ff * 3 + ii] * Woutv[ff * NVEC + gg];
    out_vec[node * NVEC * 3 + idx] = a;
  }
}

// ---------------- Host-side launch ----------------
extern "C" void kernel_launch(void* const* d_in, const int* in_sizes, int n_in,
                              void* d_out, int out_size, void* d_ws, size_t ws_size,
                              hipStream_t stream) {
  const float* x     = (const float*)d_in[0];
  const int*   snd   = (const int*)d_in[1];
  const int*   rcv   = (const int*)d_in[2];
  const float* embed = (const float*)d_in[3];
  const float* Wr1   = (const float*)d_in[4];
  const float* Wr2   = (const float*)d_in[5];
  const float* Wm    = (const float*)d_in[6];
  const float* Wc    = (const float*)d_in[7];
  const float* Wself = (const float*)d_in[8];
  const float* Wsv   = (const float*)d_in[9];
  const float* Wvv   = (const float*)d_in[10];
  const float* Wv0   = (const float*)d_in[11];
  const float* Wouts = (const float*)d_in[12];
  const float* Woutv = (const float*)d_in[13];
  const float* Wfin  = (const float*)d_in[14];
  const float* bfin  = (const float*)d_in[15];
  (void)in_sizes; (void)n_in; (void)out_size; (void)ws_size;

  // Workspace layout (all offsets 256B aligned)
  char* ws = (char*)d_ws;
  float* h_s  = (float*)(ws + 0);          // 4*512*64  f32 = 512 KB
  float* h_v  = (float*)(ws + 524288);     // 4*512*96  f32 = 768 KB
  float* a0   = (float*)(ws + 1310720);    // 512 KB
  float* a1   = (float*)(ws + 1835008);    // 768 KB
  bf16*  wfrg = (bf16*)(ws + 2621440);     // 2*23552 bf16 = 92 KB

  prep_weights_kernel<<<dim3(12), dim3(256), 0, stream>>>(Wr1, Wr2, Wm, Wsv, Wvv, Wv0, wfrg);
  init_state_kernel<<<dim3((BATCH * NNODES * F0 + 255) / 256), dim3(256), 0, stream>>>(embed, h_s);
  hipMemsetAsync(h_v, 0, (size_t)BATCH * NNODES * F1 * 3 * sizeof(float), stream);

  for (int l = 0; l < L_LAYERS; ++l) {
    hipMemsetAsync(a0, 0, (size_t)BATCH * NNODES * F0 * sizeof(float), stream);
    hipMemsetAsync(a1, 0, (size_t)BATCH * NNODES * F1 * 3 * sizeof(float), stream);
    edge_kernel<<<dim3(96, BATCH), dim3(WPB * 32), 0, stream>>>(x, snd, rcv, h_s, h_v,
                                                                wfrg, a0, a1, l);
    node_update_kernel<<<dim3(BATCH * NNODES), dim3(64), 0, stream>>>(Wc, Wself, h_s, h_v,
                                                                      a0, a1, l);
  }

  float* out_vec = (float*)d_out;                       // (B,N,16,3) first (tuple order)
  float* out_inv = out_vec + (size_t)BATCH * NNODES * NVEC * 3;
  output_kernel<<<dim3(BATCH * NNODES), dim3(64), 0, stream>>>(h_s, h_v, Wouts, Woutv,
                                                               Wfin, bfin, out_vec, out_inv);
}
